// IQARegression_50551765074566
// MI455X (gfx1250) — matrix operand order repaired and checked by reading, not compiled, not run on hardware
//
#include <hip/hip_runtime.h>
#include <hip/hip_bf16.h>
#include <math.h>

// ============================================================================
// MI455X (gfx1250) implementation.
// Heavy GEMMs (convs as GEMM/im2col, all linear projections) run through
// v_wmma_f32_16x16x32_f16 (fp16 A/B, fp32 accum). Weights are pre-transposed
// to [N,K] f16 so both A and B WMMA fragments are contiguous 128-bit loads
// matching the CDNA5 ISA 16-bit A(16x32)/B(32x16) lane layouts.
// Block tile 128x128 (8 waves, each 32x64 = 2x4 WMMA tiles) to halve A-panel
// re-streaming from HBM (im2col matrix is 226MB > 192MB L2).
// Small ops (LN, softmax-attention M=77, pooling, experts) are scalar fp32.
// ============================================================================

typedef __attribute__((ext_vector_type(16))) _Float16 v16h;
typedef __attribute__((ext_vector_type(8)))  _Float16 v8h;
typedef __attribute__((ext_vector_type(8)))  float    v8f;

#define LN_EPS 1e-5f

// ---------------------------------------------------------------------------
// Generic WMMA GEMM:  out[M,N] = A[M,K] * BT[N,K]^T  (+bias)(+relu)(accum)
// Block: 256 threads = 8 waves, block tile 128(M) x 128(N).
// Wave tile 32x64 = 2x4 WMMA 16x16 tiles. M padded to 128, N % 128 == 0.
// ---------------------------------------------------------------------------
__global__ __launch_bounds__(256) void k_gemm_wmma(
    const _Float16* __restrict__ A,   // [Mp,K] row-major f16
    const _Float16* __restrict__ BT,  // [N,K]  row-major f16 (i.e. B^T)
    float*          __restrict__ outF, // [Mp,N] f32 or null
    _Float16*       __restrict__ outH, // [Mp,N] f16 or null
    const float*    __restrict__ bias, // [N] or null
    int K, int N, int relu, int accum)
{
  const int lane = threadIdx.x & 31;
  const int wave = threadIdx.x >> 5;
  const int l16  = lane & 15;
  const int hi   = lane >> 4;
  const int m0 = blockIdx.x * 128 + (wave >> 1) * 32;
  const int n0 = blockIdx.y * 128 + (wave & 1) * 64;

  v8f acc[2][4] = {};

  const _Float16* Arow[2];
  Arow[0] = A + (size_t)(m0      + l16) * K;
  Arow[1] = A + (size_t)(m0 + 16 + l16) * K;
  const _Float16* Brow[4];
  #pragma unroll
  for (int j = 0; j < 4; ++j) Brow[j] = BT + (size_t)(n0 + j * 16 + l16) * K;

  for (int k0 = 0; k0 < K; k0 += 32) {
    // A fragment: halfs 0..7 = K[k0+hi*8 ..], halfs 8..15 = K[k0+16+hi*8 ..]
    v16h a[2], b[4];
    #pragma unroll
    for (int i = 0; i < 2; ++i) {
      v8h p0 = *(const v8h*)(Arow[i] + k0 + hi * 8);
      v8h p1 = *(const v8h*)(Arow[i] + k0 + 16 + hi * 8);
      #pragma unroll
      for (int t = 0; t < 8; ++t) { a[i][t] = p0[t]; a[i][t + 8] = p1[t]; }
    }
    // B fragment: lane gives N column; halfs 0..7 = K[k0+hi*8 ..], halfs 8..15 = K[k0+16+hi*8 ..]
    #pragma unroll
    for (int j = 0; j < 4; ++j) {
      v8h q0 = *(const v8h*)(Brow[j] + k0 + hi * 8);
      v8h q1 = *(const v8h*)(Brow[j] + k0 + 16 + hi * 8);
      #pragma unroll
      for (int t = 0; t < 8; ++t) { b[j][t] = q0[t]; b[j][t + 8] = q1[t]; }
    }

    #pragma unroll
    for (int i = 0; i < 2; ++i)
      #pragma unroll
      for (int j = 0; j < 4; ++j)
        acc[i][j] = __builtin_amdgcn_wmma_f32_16x16x32_f16(
            false, a[i], false, b[j], (short)0, acc[i][j], false, false);
  }

  // C/D layout: lane -> col (l16), VGPR r -> row = r + 8*hi within the tile
  #pragma unroll
  for (int i = 0; i < 2; ++i) {
    #pragma unroll
    for (int j = 0; j < 4; ++j) {
      const int col = n0 + j * 16 + l16;
      const float bv = bias ? bias[col] : 0.f;
      #pragma unroll
      for (int r = 0; r < 8; ++r) {
        const int row = m0 + i * 16 + hi * 8 + r;
        float v = acc[i][j][r] + bv;
        if (relu) v = fmaxf(v, 0.f);
        const size_t idx = (size_t)row * N + col;
        if (outF) { if (accum) outF[idx] += v; else outF[idx] = v; }
        if (outH) outH[idx] = (_Float16)v;
      }
    }
  }
}

// ---------------------------------------------------------------------------
// Conversion / layout kernels
// ---------------------------------------------------------------------------
__global__ void k_cvt_f16(const float* __restrict__ in, _Float16* __restrict__ out, int n) {
  int i = blockIdx.x * 256 + threadIdx.x;
  if (i < n) out[i] = (_Float16)in[i];
}

// in [K,N] f32 -> out [N,K] f16
__global__ void k_transpose_f16(const float* __restrict__ in, _Float16* __restrict__ out,
                                int K, int N) {
  int t = blockIdx.x * 256 + threadIdx.x;
  if (t < K * N) {
    int n = t / K, k = t - n * K;
    out[(size_t)n * K + k] = (_Float16)in[(size_t)k * N + n];
  }
}

// x [64,3072,16,16] -> A1 [16384,3072] f16, A1[(b*256+s)*3072+c] = x[(b*3072+c)*256+s]
__global__ void k_x_to_a1(const float* __restrict__ x, _Float16* __restrict__ out) {
  size_t t = (size_t)blockIdx.x * 256 + threadIdx.x;
  const size_t total = (size_t)16384 * 3072;
  if (t < total) {
    int c = (int)(t % 3072);
    int m = (int)(t / 3072);
    int b = m >> 8, s = m & 255;
    out[t] = (_Float16)x[((size_t)b * 3072 + c) * 256 + s];
  }
}

// f16 feature map [16384 rows(b,s), 768] -> im2col [16384, 6912], K idx = c*9 + r*3 + s2
__global__ void k_im2col(const _Float16* __restrict__ f, _Float16* __restrict__ out) {
  int t = blockIdx.x * 256 + threadIdx.x;   // over 16384*768
  if (t < 16384 * 768) {
    int c = t % 768;
    int m = t / 768;
    int b = m >> 8, sp = m & 255;
    int d2 = sp >> 4, d3 = sp & 15;
    _Float16* o = out + (size_t)m * 6912 + c * 9;
    #pragma unroll
    for (int r = 0; r < 3; ++r)
      #pragma unroll
      for (int s2 = 0; s2 < 3; ++s2) {
        int y = d2 + r - 1, xx = d3 + s2 - 1;
        _Float16 v = (_Float16)0.f;
        if (y >= 0 && y < 16 && xx >= 0 && xx < 16)
          v = f[(size_t)(b * 256 + y * 16 + xx) * 768 + c];
        o[r * 3 + s2] = v;
      }
  }
}

// ---------------------------------------------------------------------------
// LayerNorm over last dim (512), one wave per row.
// ---------------------------------------------------------------------------
__global__ __launch_bounds__(256) void k_ln(
    const float* __restrict__ in, const float* __restrict__ g, const float* __restrict__ bta,
    _Float16* __restrict__ outH, float* __restrict__ outF, int outStride, int rows)
{
  int row  = (int)((blockIdx.x * 256 + threadIdx.x) >> 5);
  int lane = threadIdx.x & 31;
  if (row >= rows) return;
  const float* r = in + (size_t)row * 512;
  float v[16];
  float s = 0.f;
  #pragma unroll
  for (int i = 0; i < 16; ++i) { v[i] = r[lane + i * 32]; s += v[i]; }
  for (int o = 16; o > 0; o >>= 1) s += __shfl_xor(s, o, 32);
  float mean = s * (1.f / 512.f);
  float var = 0.f;
  #pragma unroll
  for (int i = 0; i < 16; ++i) { float d = v[i] - mean; var += d * d; }
  for (int o = 16; o > 0; o >>= 1) var += __shfl_xor(var, o, 32);
  float rstd = rsqrtf(var * (1.f / 512.f) + LN_EPS);
  #pragma unroll
  for (int i = 0; i < 16; ++i) {
    int c = lane + i * 32;
    float o = (v[i] - mean) * rstd * g[c] + bta[c];
    if (outH) outH[(size_t)row * 512 + c] = (_Float16)o;
    if (outF) outF[(size_t)row * outStride + c] = o;
  }
}

// ---------------------------------------------------------------------------
// Scalar MHA: heads=8, dh=64. One wave per (b,h,query). M <= 77 keys.
// Q rows = b*N+i, K/V rows = b*M+j, head h occupies cols [h*64, h*64+64).
// ---------------------------------------------------------------------------
__global__ __launch_bounds__(256) void k_attn(
    const float* __restrict__ Q, const float* __restrict__ K, const float* __restrict__ V,
    float* __restrict__ O, int B, int N, int M)
{
  __shared__ float sc[8][128];
  const int wv = threadIdx.x >> 5, lane = threadIdx.x & 31;
  const int gq = blockIdx.x * 8 + wv;
  const int total = B * 8 * N;
  const bool active = gq < total;
  int b = 0, h = 0, i = 0;
  if (active) { b = gq / (8 * N); int rem = gq - b * 8 * N; h = rem / N; i = rem - h * N; }
  const float* qrow = Q + ((size_t)(b * N + i) * 512 + h * 64);

  if (active) {
    for (int j = lane; j < M; j += 32) {
      const float* krow = K + ((size_t)(b * M + j) * 512 + h * 64);
      float s = 0.f;
      #pragma unroll 8
      for (int d = 0; d < 64; ++d) s += qrow[d] * krow[d];
      sc[wv][j] = s * 0.125f;   // 64^-0.5
    }
  }
  __syncthreads();
  float mx = -1e30f, sum = 0.f;
  if (active) {
    for (int j = 0; j < M; ++j) mx = fmaxf(mx, sc[wv][j]);
    for (int j = 0; j < M; ++j) sum += __expf(sc[wv][j] - mx);
  }
  __syncthreads();
  if (active) {
    float inv = 1.f / sum;
    for (int j = lane; j < M; j += 32) sc[wv][j] = __expf(sc[wv][j] - mx) * inv;
  }
  __syncthreads();
  if (active) {
    float o0 = 0.f, o1 = 0.f;
    for (int j = 0; j < M; ++j) {
      float p = sc[wv][j];
      const float* vrow = V + ((size_t)(b * M + j) * 512 + h * 64);
      o0 += p * vrow[lane];
      o1 += p * vrow[lane + 32];
    }
    float* orow = O + ((size_t)(b * N + i) * 512 + h * 64);
    orow[lane] = o0; orow[lane + 32] = o1;
  }
}

// ---------------------------------------------------------------------------
// IQAPooling helpers
// ---------------------------------------------------------------------------
// quality weight: sigmoid(sum_c f[row,c]*w[c] + b). One wave per row (b*256+s).
__global__ __launch_bounds__(256) void k_qualw(
    const float* __restrict__ f, const float* __restrict__ w, const float* __restrict__ bscal,
    float* __restrict__ qw, int rows)
{
  int row  = (int)((blockIdx.x * 256 + threadIdx.x) >> 5);
  int lane = threadIdx.x & 31;
  if (row >= rows) return;
  const float* r = f + (size_t)row * 512;
  float s = 0.f;
  #pragma unroll
  for (int i = 0; i < 16; ++i) { int c = lane + i * 32; s += r[c] * w[c]; }
  for (int o = 16; o > 0; o >>= 1) s += __shfl_xor(s, o, 32);
  if (lane == 0) qw[row] = 1.f / (1.f + __expf(-(s + bscal[0])));
}

// channel mean over 256 spatial positions
__global__ void k_chmean(const float* __restrict__ f, float* __restrict__ cm) {
  int t = blockIdx.x * 256 + threadIdx.x;   // 64*512
  if (t < 64 * 512) {
    int b = t >> 9, c = t & 511;
    float s = 0.f;
    for (int sp = 0; sp < 256; ++sp) s += f[(size_t)(b * 256 + sp) * 512 + c];
    cm[t] = s * (1.f / 256.f);
  }
}

// channel attention MLP 512->32(relu)->512(sigmoid). One block per batch.
__global__ __launch_bounds__(256) void k_chw(
    const float* __restrict__ cm, const float* __restrict__ w1, const float* __restrict__ b1,
    const float* __restrict__ w2, const float* __restrict__ b2, float* __restrict__ chw)
{
  __shared__ float hid[32];
  int b = blockIdx.x, tid = threadIdx.x;
  if (tid < 32) {
    float s = b1[tid];
    for (int c = 0; c < 512; ++c) s += cm[(size_t)b * 512 + c] * w1[c * 32 + tid];
    hid[tid] = fmaxf(s, 0.f);
  }
  __syncthreads();
  for (int c = tid; c < 512; c += 256) {
    float s = b2[c];
    #pragma unroll
    for (int h = 0; h < 32; ++h) s += hid[h] * w2[h * 512 + c];
    chw[(size_t)b * 512 + c] = 1.f / (1.f + __expf(-s));
  }
}

// csf 224x224 -> 16x16 (jax.image.resize bilinear with antialias triangle, r=14)
// plus per-batch sum. One block (256 threads) per batch.
__global__ __launch_bounds__(256) void k_csf(
    const float* __restrict__ csf, float* __restrict__ out, float* __restrict__ sums)
{
  __shared__ float red[256];
  int b = blockIdx.x, t = threadIdx.x;
  int d2 = t >> 4, d3 = t & 15;
  const float* img = csf + (size_t)b * 224 * 224;
  float cy = (d2 + 0.5f) * 14.f - 0.5f;
  float cx = (d3 + 0.5f) * 14.f - 0.5f;
  int y0 = (int)ceilf(cy - 14.f);  if (y0 < 0) y0 = 0;
  int y1 = (int)floorf(cy + 14.f); if (y1 > 223) y1 = 223;
  int x0 = (int)ceilf(cx - 14.f);  if (x0 < 0) x0 = 0;
  int x1 = (int)floorf(cx + 14.f); if (x1 > 223) x1 = 223;
  float sy = 0.f, sx = 0.f;
  for (int y = y0; y <= y1; ++y) sy += fmaxf(0.f, 1.f - fabsf(y - cy) * (1.f / 14.f));
  for (int x = x0; x <= x1; ++x) sx += fmaxf(0.f, 1.f - fabsf(x - cx) * (1.f / 14.f));
  float val = 0.f;
  for (int y = y0; y <= y1; ++y) {
    float wy = fmaxf(0.f, 1.f - fabsf(y - cy) * (1.f / 14.f));
    const float* rowp = img + y * 224;
    float acc = 0.f;
    for (int x = x0; x <= x1; ++x)
      acc += fmaxf(0.f, 1.f - fabsf(x - cx) * (1.f / 14.f)) * rowp[x];
    val += wy * acc;
  }
  val /= (sy * sx);
  out[(size_t)b * 256 + t] = val;
  red[t] = val;
  __syncthreads();
  for (int o = 128; o > 0; o >>= 1) { if (t < o) red[t] += red[t + o]; __syncthreads(); }
  if (t == 0) sums[b] = red[0];
}

// weighted pooling -> fcat[b, 0..511]
__global__ void k_pool(const float* __restrict__ f, const float* __restrict__ qw,
    const float* __restrict__ chw, const float* __restrict__ csf16,
    const float* __restrict__ csfsum, float* __restrict__ fcat)
{
  int t = blockIdx.x * 256 + threadIdx.x;   // 64*512
  if (t < 64 * 512) {
    int b = t >> 9, c = t & 511;
    float inv = 1.f / csfsum[b];
    float s = 0.f;
    for (int sp = 0; sp < 256; ++sp) {
      float w = csf16[(size_t)b * 256 + sp] * inv + 0.4f;
      s += f[(size_t)(b * 256 + sp) * 512 + c] * qw[b * 256 + sp] * w;
    }
    fcat[(size_t)b * 1024 + c] = s * chw[t];
  }
}

// proj_pool over sequence: [B,35,512] x pp_w[35] + pp_b -> [B,512]
__global__ void k_conspool(const float* __restrict__ cons, const float* __restrict__ w,
    const float* __restrict__ bs, float* __restrict__ out)
{
  int t = blockIdx.x * 256 + threadIdx.x;   // 64*512
  if (t < 64 * 512) {
    int b = t >> 9, c = t & 511;
    float s = bs[0];
    #pragma unroll
    for (int n = 0; n < 35; ++n) s += cons[(size_t)(b * 35 + n) * 512 + c] * w[n];
    out[(size_t)b * 512 + c] = s;
  }
}

// dense expert compute: epred[b,e] = h(b,e,:) . ew2[e,:] + eb2[e],
// h = gelu(fcat[b,:] @ ew1[e]). One block per (b,e).
__global__ __launch_bounds__(256) void k_expert(
    const float* __restrict__ fcat, const float* __restrict__ ew1,
    const float* __restrict__ eb1, const float* __restrict__ ew2,
    const float* __restrict__ eb2, float* __restrict__ epred)
{
  __shared__ float xf[1024];
  __shared__ float red[256];
  int be = blockIdx.x, b = be >> 2, e = be & 3, tid = threadIdx.x;
  for (int i = tid; i < 1024; i += 256) xf[i] = fcat[(size_t)b * 1024 + i];
  __syncthreads();
  float acc = 0.f;
  for (int o = tid; o < 512; o += 256) {
    float d = eb1[e * 512 + o];
    const float* w = ew1 + (size_t)e * 1024 * 512 + o;
    for (int i = 0; i < 1024; ++i) d += xf[i] * w[(size_t)i * 512];
    float gg = 0.5f * d * (1.f + erff(d * 0.70710678118654752f));
    acc += gg * ew2[e * 512 + o];
  }
  red[tid] = acc;
  __syncthreads();
  for (int o = 128; o > 0; o >>= 1) { if (tid < o) red[tid] += red[tid + o]; __syncthreads(); }
  if (tid == 0) epred[be] = red[0] + eb2[e];
}

// gate softmax + top-3 + combine. 64 threads, one per batch.
__global__ void k_final(const float* __restrict__ fcat, const float* __restrict__ gw,
    const float* __restrict__ gb, const float* __restrict__ epred, float* __restrict__ out)
{
  int b = threadIdx.x;
  if (b < 64) {
    float lg[4];
    for (int e = 0; e < 4; ++e) {
      float s = gb[e];
      for (int i = 0; i < 1024; ++i) s += fcat[(size_t)b * 1024 + i] * gw[i * 4 + e];
      lg[e] = s;
    }
    float m = fmaxf(fmaxf(lg[0], lg[1]), fmaxf(lg[2], lg[3]));
    float p[4]; float sum = 0.f;
    for (int e = 0; e < 4; ++e) { p[e] = __expf(lg[e] - m); sum += p[e]; }
    for (int e = 0; e < 4; ++e) p[e] /= sum;
    bool used[4] = {false, false, false, false};
    float pred = 0.f;
    for (int rk = 0; rk < 3; ++rk) {
      int best = 0; float bv = -1e30f;
      for (int e = 0; e < 4; ++e) if (!used[e] && p[e] > bv) { bv = p[e]; best = e; }
      used[best] = true;
      pred += bv * epred[b * 4 + best];
    }
    out[b] = pred;
  }
}

// ===========================================================================
extern "C" void kernel_launch(void* const* d_in, const int* in_sizes, int n_in,
                              void* d_out, int out_size, void* d_ws, size_t ws_size,
                              hipStream_t stream) {
  // ---- inputs ----
  const float* x        = (const float*)d_in[0];   // [64,3072,16,16]
  const float* text     = (const float*)d_in[1];   // [64,77,768]
  const float* texta    = (const float*)d_in[2];   // [64,35,768]
  const float* csf      = (const float*)d_in[3];   // [64,1,224,224]
  const float* dc_w     = (const float*)d_in[4];   // [768,3072]
  const float* dc_b     = (const float*)d_in[5];
  const float* conv_w   = (const float*)d_in[6];   // [512,768,3,3]
  const float* conv_b   = (const float*)d_in[7];
  const float* proj     = (const float*)d_in[8];   // [768,512]
  const float* ln1_g    = (const float*)d_in[9];
  const float* ln1_b    = (const float*)d_in[10];
  const float* ln2_g    = (const float*)d_in[11];
  const float* ln2_b    = (const float*)d_in[12];
  const float* ln3_g    = (const float*)d_in[13];
  const float* ln3_b    = (const float*)d_in[14];
  const float* ca_wq    = (const float*)d_in[15];
  const float* ca_wk    = (const float*)d_in[16];
  const float* ca_wv    = (const float*)d_in[17];
  const float* ca_wo    = (const float*)d_in[18];
  const float* ca_bo    = (const float*)d_in[19];
  const float* cat_wq   = (const float*)d_in[20];
  const float* cat_wk   = (const float*)d_in[21];
  const float* cat_wv   = (const float*)d_in[22];
  const float* cat_wo   = (const float*)d_in[23];
  const float* cat_bo   = (const float*)d_in[24];
  const float* pp_w     = (const float*)d_in[25];
  const float* pp_b     = (const float*)d_in[26];
  const float* pcw      = (const float*)d_in[27];
  const float* pcb      = (const float*)d_in[28];
  const float* ca1_w    = (const float*)d_in[29];
  const float* ca1_b    = (const float*)d_in[30];
  const float* ca2_w    = (const float*)d_in[31];
  const float* ca2_b    = (const float*)d_in[32];
  const float* gate_w   = (const float*)d_in[33];
  const float* gate_b   = (const float*)d_in[34];
  const float* ew1      = (const float*)d_in[35];
  const float* eb1      = (const float*)d_in[36];
  const float* ew2      = (const float*)d_in[37];
  const float* eb2      = (const float*)d_in[38];
  float* out = (float*)d_out;

  // ---- workspace allocator (256B aligned) ----
  size_t off = 0;
  auto alloc = [&](size_t bytes) -> char* {
    char* p = (char*)d_ws + off;
    off += (bytes + 255) & ~(size_t)255;
    return p;
  };
  const int MP_IMG = 16384;          // 64*256 (multiple of 128)
  const int MP_TF  = 4992;           // pad of 64*77=4928
  const int MP_TFA = 2304;           // pad of 64*35=2240

  _Float16* A1     = (_Float16*)alloc((size_t)MP_IMG * 3072 * 2);
  _Float16* W1     = (_Float16*)alloc((size_t)768 * 3072 * 2);
  _Float16* W2     = (_Float16*)alloc((size_t)512 * 6912 * 2);
  _Float16* PROJT  = (_Float16*)alloc((size_t)512 * 768 * 2);
  _Float16* WT[8];
  for (int i = 0; i < 8; ++i) WT[i] = (_Float16*)alloc((size_t)512 * 512 * 2);
  _Float16* F1H    = (_Float16*)alloc((size_t)MP_IMG * 768 * 2);
  _Float16* A2     = (_Float16*)alloc((size_t)MP_IMG * 6912 * 2);
  float*    FDIS   = (float*)   alloc((size_t)MP_IMG * 512 * 4);
  _Float16* TXT1H  = (_Float16*)alloc((size_t)MP_TF * 768 * 2);
  _Float16* TXTA1H = (_Float16*)alloc((size_t)MP_TFA * 768 * 2);
  float*    TF     = (float*)   alloc((size_t)MP_TF * 512 * 4);
  float*    TFA    = (float*)   alloc((size_t)MP_TFA * 512 * 4);
  _Float16* TFH    = (_Float16*)alloc((size_t)MP_TF * 512 * 2);
  _Float16* TFAH   = (_Float16*)alloc((size_t)MP_TFA * 512 * 2);
  _Float16* LN1H   = (_Float16*)alloc((size_t)MP_IMG * 512 * 2);
  float*    FDISR  = (float*)   alloc((size_t)MP_IMG * 512 * 4);
  _Float16* LN2H   = (_Float16*)alloc((size_t)MP_TF * 512 * 2);
  float*    Qb     = (float*)   alloc((size_t)MP_IMG * 512 * 4);
  float*    KC     = (float*)   alloc((size_t)MP_TF * 512 * 4);
  float*    VC     = (float*)   alloc((size_t)MP_TF * 512 * 4);
  float*    AO     = (float*)   alloc((size_t)MP_IMG * 512 * 4);
  _Float16* AOH    = (_Float16*)alloc((size_t)MP_IMG * 512 * 2);
  float*    Q2     = (float*)   alloc((size_t)MP_TFA * 512 * 4);
  float*    K2     = (float*)   alloc((size_t)MP_TF * 512 * 4);
  float*    V2     = (float*)   alloc((size_t)MP_TF * 512 * 4);
  float*    AO2    = (float*)   alloc((size_t)MP_TFA * 512 * 4);
  _Float16* AO2H   = (_Float16*)alloc((size_t)MP_TFA * 512 * 2);
  float*    CONS   = (float*)   alloc((size_t)MP_TFA * 512 * 4);
  float*    CONSP  = (float*)   alloc((size_t)64 * 512 * 4);
  float*    QW     = (float*)   alloc((size_t)16384 * 4);
  float*    CHM    = (float*)   alloc((size_t)64 * 512 * 4);
  float*    CHW    = (float*)   alloc((size_t)64 * 512 * 4);
  float*    CSF16  = (float*)   alloc((size_t)64 * 256 * 4);
  float*    CSFSUM = (float*)   alloc((size_t)64 * 4);
  float*    FCAT   = (float*)   alloc((size_t)64 * 1024 * 4);
  float*    EPRED  = (float*)   alloc((size_t)64 * 4 * 4);
  (void)ws_size; (void)in_sizes; (void)n_in; (void)out_size;

  auto blocks = [](size_t n) { return (unsigned)((n + 255) / 256); };

  // ---- weight prep ----
  k_cvt_f16<<<blocks((size_t)768 * 3072), 256, 0, stream>>>(dc_w, W1, 768 * 3072);
  k_cvt_f16<<<blocks((size_t)512 * 6912), 256, 0, stream>>>(conv_w, W2, 512 * 6912);
  k_transpose_f16<<<blocks((size_t)768 * 512), 256, 0, stream>>>(proj, PROJT, 768, 512);
  const float* wsrc[8] = {ca_wq, ca_wk, ca_wv, ca_wo, cat_wq, cat_wk, cat_wv, cat_wo};
  for (int i = 0; i < 8; ++i)
    k_transpose_f16<<<blocks((size_t)512 * 512), 256, 0, stream>>>(wsrc[i], WT[i], 512, 512);

  // ---- activation prep ----
  k_x_to_a1<<<blocks((size_t)MP_IMG * 3072), 256, 0, stream>>>(x, A1);
  k_cvt_f16<<<blocks((size_t)4928 * 768), 256, 0, stream>>>(text, TXT1H, 4928 * 768);
  k_cvt_f16<<<blocks((size_t)2240 * 768), 256, 0, stream>>>(texta, TXTA1H, 2240 * 768);

  // ---- conv1 (1x1) : [16384,3072] x [3072->768] ----
  k_gemm_wmma<<<dim3(MP_IMG / 128, 768 / 128), 256, 0, stream>>>(
      A1, W1, nullptr, F1H, dc_b, 3072, 768, 0, 0);
  // ---- conv2 (3x3) via im2col : K=6912 -> 512, +bias +relu ----
  k_im2col<<<blocks((size_t)16384 * 768), 256, 0, stream>>>(F1H, A2);
  k_gemm_wmma<<<dim3(MP_IMG / 128, 512 / 128), 256, 0, stream>>>(
      A2, W2, FDIS, nullptr, conv_b, 6912, 512, 1, 0);

  // ---- text projections ----
  k_gemm_wmma<<<dim3(MP_TF / 128, 512 / 128), 256, 0, stream>>>(
      TXT1H, PROJT, TF, nullptr, nullptr, 768, 512, 0, 0);
  k_gemm_wmma<<<dim3(MP_TFA / 128, 512 / 128), 256, 0, stream>>>(
      TXTA1H, PROJT, TFA, nullptr, nullptr, 768, 512, 0, 0);

  // ---- layernorms + f16 copies ----
  k_ln<<<blocks((size_t)16384 * 32), 256, 0, stream>>>(FDIS, ln1_g, ln1_b, LN1H, FDISR, 512, 16384);
  k_ln<<<blocks((size_t)4928 * 32), 256, 0, stream>>>(TF, ln2_g, ln2_b, LN2H, nullptr, 512, 4928);
  k_cvt_f16<<<blocks((size_t)4928 * 512), 256, 0, stream>>>(TF, TFH, 4928 * 512);
  k_cvt_f16<<<blocks((size_t)2240 * 512), 256, 0, stream>>>(TFA, TFAH, 2240 * 512);

  // ---- cross-attention (image queries -> ln2(tf)) ----
  k_gemm_wmma<<<dim3(MP_IMG / 128, 4), 256, 0, stream>>>(LN1H, WT[0], Qb, nullptr, nullptr, 512, 512, 0, 0);
  k_gemm_wmma<<<dim3(MP_TF / 128, 4), 256, 0, stream>>>(LN2H, WT[1], KC, nullptr, nullptr, 512, 512, 0, 0);
  k_gemm_wmma<<<dim3(MP_TF / 128, 4), 256, 0, stream>>>(LN2H, WT[2], VC, nullptr, nullptr, 512, 512, 0, 0);
  k_attn<<<(64 * 8 * 256) / 8, 256, 0, stream>>>(Qb, KC, VC, AO, 64, 256, 77);
  k_cvt_f16<<<blocks((size_t)16384 * 512), 256, 0, stream>>>(AO, AOH, 16384 * 512);
  // output proj + residual add onto ln1(f_dis)
  k_gemm_wmma<<<dim3(MP_IMG / 128, 4), 256, 0, stream>>>(AOH, WT[3], FDISR, nullptr, ca_bo, 512, 512, 0, 1);

  // ---- cat attention (tfa queries -> tf) ----
  k_gemm_wmma<<<dim3(MP_TFA / 128, 4), 256, 0, stream>>>(TFAH, WT[4], Q2, nullptr, nullptr, 512, 512, 0, 0);
  k_gemm_wmma<<<dim3(MP_TF / 128, 4), 256, 0, stream>>>(TFH, WT[5], K2, nullptr, nullptr, 512, 512, 0, 0);
  k_gemm_wmma<<<dim3(MP_TF / 128, 4), 256, 0, stream>>>(TFH, WT[6], V2, nullptr, nullptr, 512, 512, 0, 0);
  k_attn<<<(64 * 8 * 35 + 7) / 8, 256, 0, stream>>>(Q2, K2, V2, AO2, 64, 35, 77);
  k_cvt_f16<<<blocks((size_t)2240 * 512), 256, 0, stream>>>(AO2, AO2H, 2240 * 512);
  k_gemm_wmma<<<dim3(MP_TFA / 128, 4), 256, 0, stream>>>(AO2H, WT[7], CONS, nullptr, cat_bo, 512, 512, 0, 0);

  // ---- cons pooling + ln3 (writes high half of FCAT) ----
  k_conspool<<<blocks((size_t)64 * 512), 256, 0, stream>>>(CONS, pp_w, pp_b, CONSP);
  k_ln<<<blocks((size_t)64 * 32), 256, 0, stream>>>(CONSP, ln3_g, ln3_b, nullptr, FCAT + 512, 1024, 64);

  // ---- IQA pooling (writes low half of FCAT) ----
  k_csf<<<64, 256, 0, stream>>>(csf, CSF16, CSFSUM);
  k_qualw<<<blocks((size_t)16384 * 32), 256, 0, stream>>>(FDISR, pcw, pcb, QW, 16384);
  k_chmean<<<blocks((size_t)64 * 512), 256, 0, stream>>>(FDISR, CHM);
  k_chw<<<64, 256, 0, stream>>>(CHM, ca1_w, ca1_b, ca2_w, ca2_b, CHW);
  k_pool<<<blocks((size_t)64 * 512), 256, 0, stream>>>(FDISR, QW, CHW, CSF16, CSFSUM, FCAT);

  // ---- gate + experts ----
  k_expert<<<256, 256, 0, stream>>>(FCAT, ew1, eb1, ew2, eb2, EPRED);
  k_final<<<1, 64, 0, stream>>>(FCAT, gate_w, gate_b, EPRED, out);
}
